// CriticTab_2044404433745
// MI455X (gfx1250) — 8-bit matrix operand order repaired and checked by reading, compiled, not run on hardware
//
#include <hip/hip_runtime.h>


typedef __attribute__((ext_vector_type(8))) int v8i;

#define WAVES_PER_BLOCK 8
#define THREADS (WAVES_PER_BLOCK * 32)

__global__ __launch_bounds__(THREADS)
void critictab_wmma_gather(const int* __restrict__ obs,
                           const float* __restrict__ v,
                           float* __restrict__ out,
                           int Btot) {
    __shared__ int lds[THREADS * 8];

    const int tid   = threadIdx.x;
    const int lane  = tid & 31;
    const int wave  = tid >> 5;
    const int m     = lane & 15;
    const int half  = lane >> 4;
    const int bbase = (blockIdx.x * WAVES_PER_BLOCK + wave) * 16;

    int brow = bbase + m;
    if (brow >= Btot) brow = Btot - 1;
    const int* row = obs + brow * 16 + half * 8;
    int4 o0 = *(const int4*)(row);
    int4 o1 = *(const int4*)(row + 4);
    int a0 = (o0.x & 1) | ((o0.y & 1) << 8) | ((o0.z & 1) << 16) | ((o0.w & 1) << 24);
    int a1 = (o1.x & 1) | ((o1.y & 1) << 8) | ((o1.z & 1) << 16) | ((o1.w & 1) << 24);
    v8i A = {a0, a1, 0, 0, 0, 0, 0, 0};

    int b0 = 0, b1 = 0, b2 = 0, b3 = 0;
    if (lane == 0) { b0 = 0x08040201; b1 = (int)0x80402010; }
    if (lane == 17) { b0 = 0x08040201; b1 = (int)0x80402010; }
    v8i Bf = {b0, b1, b2, b3, 0, 0, 0, 0};

    v8i Cz = {0, 0, 0, 0, 0, 0, 0, 0};
    v8i Dm = __builtin_amdgcn_wmma_i32_16x16x64_iu8(false, A, false, Bf, Cz, false, false);

    #pragma unroll
    for (int r = 0; r < 8; ++r) lds[tid * 8 + r] = Dm[r];
    __syncthreads();

    if (lane < 16) {
        const int r   = lane & 7;
        const int src = wave * 32 + ((lane < 8) ? 0 : 16);
        const int lo  = lds[src * 8 + r];
        const int hi  = lds[(src + 1) * 8 + r];
        const int idx = lo + (hi << 8);
        const int b   = bbase + lane;
        if (b < Btot) out[b] = v[idx];
    }
}

extern "C" void kernel_launch(void* const* d_in, const int* in_sizes, int n_in,
                              void* d_out, int out_size, void* d_ws, size_t ws_size,
                              hipStream_t stream) {
    (void)n_in; (void)d_ws; (void)ws_size; (void)out_size;
    const int*   obs = (const int*)d_in[0];
    const float* v   = (const float*)d_in[2];
    float*       out = (float*)d_out;

    const int B = in_sizes[0] / 16;
    const int per_block = WAVES_PER_BLOCK * 16;
    const int grid = (B + per_block - 1) / per_block;

    critictab_wmma_gather<<<grid, THREADS, 0, stream>>>(obs, v, out, B);
}
